// Gate_51616916963810
// MI455X (gfx1250) — compile-verified, matrix operand order checked by reading, not run on hardware
//
#include <hip/hip_runtime.h>
#include <hip/hip_bf16.h>

#define DIM      7168
#define NEXP     256
#define NGROUPS  8
#define GSIZE    32           // experts per group
#define TOPKG    4
#define TOPK     8
#define MTILE    16
#define KC       256          // f32 K-elements per LDS stage (16 KB / buffer)
#define NSTAGES  (DIM / KC)   // 28
#define KCHUNKS  (DIM / 32)   // 224 WMMA k-chunks
#define CPS      (KC / 32)    // 8 chunks per stage (8*32 lanes == 256 threads)
#define ROUTE_SCALE 2.5f

typedef __attribute__((ext_vector_type(16))) __bf16   v16bf;
typedef __attribute__((ext_vector_type(8)))  float    v8f;
typedef __attribute__((ext_vector_type(4)))  unsigned ui4;
typedef __attribute__((ext_vector_type(8)))  int      i8v;
typedef __attribute__((ext_vector_type(4)))  int      i4v;

__device__ __forceinline__ unsigned short f2bf(float f) {
  unsigned u = __builtin_bit_cast(unsigned, f);
  u += 0x7FFFu + ((u >> 16) & 1u);            // round-to-nearest-even
  return (unsigned short)(u >> 16);
}

// ---------------------------------------------------------------------------
// Kernel 0: pre-swizzle weight f32[256][7168] -> bf16 WMMA B-fragments in ws.
//   ws[(((t*KCHUNKS + c)*32) + l)*16 + j]
// with  n = t*16 + (l&15),  k = c*32 + j + 8*(l>=16) + 8*(j>=8)
// (16-bit B-matrix 32x16 VGPR layout, CDNA5 ISA §7.12.2).
// ---------------------------------------------------------------------------
__global__ void __launch_bounds__(256)
swizzle_weight(const float* __restrict__ w, unsigned short* __restrict__ out) {
  int tid  = blockIdx.x * 256 + threadIdx.x;
  int j    = tid & 15;
  int l    = (tid >> 4) & 31;
  int frag = tid >> 9;
  int c    = frag % KCHUNKS;
  int t    = frag / KCHUNKS;
  int n    = t * 16 + (l & 15);
  int k    = c * 32 + j + ((l >> 4) << 3) + ((j >> 3) << 3);
  out[tid] = f2bf(w[(size_t)n * DIM + k]);
}

// ---------------------------------------------------------------------------
// TDM: async-load a 16 x KC f32 tile (row stride DIM) into LDS.
// D# packing per CDNA5 ISA (6-arg clang-23 builtin).
// ---------------------------------------------------------------------------
__device__ __forceinline__ void tdm_load_tile(unsigned lds_off, const float* gaddr) {
  unsigned long long ga = (unsigned long long)(size_t)gaddr;
  ui4 g0 = { 1u,                                              // count=1 (valid user D#)
             lds_off,                                         // lds_addr
             (unsigned)ga,                                    // global_addr[31:0]
             ((unsigned)(ga >> 32) & 0x01FFFFFFu) | 0x80000000u }; // addr[56:32] | type=2
  i8v g1 = { (int)0x00020000,                                 // wg_mask=0, data_size=4B
             (int)((DIM & 0xFFFF) << 16),                     // tensor_dim0[15:0]
             (int)((DIM >> 16) | (MTILE << 16)),              // dim0[31:16] | tensor_dim1 lo
             (int)(KC << 16),                                 // dim1 hi | tile_dim0 = KC
             MTILE,                                           // tile_dim1=16, tile_dim2=0
             DIM,                                             // tensor_dim0_stride lo
             0, 0 };
  i4v z4 = { 0, 0, 0, 0 };
  i8v z8 = { 0, 0, 0, 0, 0, 0, 0, 0 };
  __builtin_amdgcn_tensor_load_to_lds(g0, g1, z4, z4, z8, 0);
}

// ---------------------------------------------------------------------------
// Fused gate kernel: one workgroup = 16 tokens x 256 experts.
// 8 waves, wave w owns expert tiles 2w, 2w+1 (32 experts).
// ---------------------------------------------------------------------------
__global__ void __launch_bounds__(256, 1)
gate_fused(const float* __restrict__ x, const float* __restrict__ bias,
           const unsigned short* __restrict__ wfrag,
           float* __restrict__ outw, float* __restrict__ outi)
{
  __shared__ __align__(16) float          xbuf[2][MTILE * KC];       // 2 x 16 KB TDM dst
  __shared__ __align__(16) unsigned short abuf[CPS * 32 * 16];       // 8 KB bf16 A-frags
  __shared__ __align__(16) float          logits[MTILE * NEXP];      // 16 KB
  __shared__ float sbias[NEXP];

  const int tid   = threadIdx.x;
  const int lane  = tid & 31;
  const int wave  = tid >> 5;
  const int mbase = blockIdx.x * MTILE;

  sbias[tid] = bias[tid];

  const float* xrow = x + (size_t)mbase * DIM;

  if (wave == 0) {                       // prologue: fill both buffers
    tdm_load_tile((unsigned)(size_t)(void*)&xbuf[0][0], xrow + 0 * KC);
    tdm_load_tile((unsigned)(size_t)(void*)&xbuf[1][0], xrow + 1 * KC);
  }

  v8f c0 = {0.f,0.f,0.f,0.f,0.f,0.f,0.f,0.f};
  v8f c1 = {0.f,0.f,0.f,0.f,0.f,0.f,0.f,0.f};
  const int t0 = wave * 2, t1 = wave * 2 + 1;

  // Cooperative-conversion role of this thread: one fragment-lane per thread.
  const int ckk = tid >> 5;              // chunk within stage (0..7)
  const int cl  = tid & 31;              // fragment lane
  const int cm  = cl & 15;               // token row
  const int cb  = (cl >> 4) * 8;         // K sub-block select per half-wave

  for (int s = 0; s < NSTAGES; ++s) {
    if (wave == 0) {                     // TDM in-order: chunk s landed when cnt low
      if (s < NSTAGES - 1) __builtin_amdgcn_s_wait_tensorcnt(1);
      else                 __builtin_amdgcn_s_wait_tensorcnt(0);
    }
    __syncthreads();                     // xbuf[s&1] ready; abuf[s-1] fully consumed

    // ---- cooperative f32 -> bf16 conversion into A-fragment layout ----
    {
      const float* cp = &xbuf[s & 1][cm * KC + ckk * 32 + cb];
      float4 f0 = *(const float4*)(cp);
      float4 f1 = *(const float4*)(cp + 4);
      float4 f2 = *(const float4*)(cp + 16);
      float4 f3 = *(const float4*)(cp + 20);
      union { unsigned short h[16]; uint4 q[2]; } T;
      T.h[0]=f2bf(f0.x);  T.h[1]=f2bf(f0.y);  T.h[2]=f2bf(f0.z);  T.h[3]=f2bf(f0.w);
      T.h[4]=f2bf(f1.x);  T.h[5]=f2bf(f1.y);  T.h[6]=f2bf(f1.z);  T.h[7]=f2bf(f1.w);
      T.h[8]=f2bf(f2.x);  T.h[9]=f2bf(f2.y);  T.h[10]=f2bf(f2.z); T.h[11]=f2bf(f2.w);
      T.h[12]=f2bf(f3.x); T.h[13]=f2bf(f3.y); T.h[14]=f2bf(f3.z); T.h[15]=f2bf(f3.w);
      uint4* ap = (uint4*)&abuf[(ckk * 32 + cl) * 16];
      ap[0] = T.q[0];
      ap[1] = T.q[1];
    }
    __syncthreads();                     // abuf ready; xbuf[s&1] is now free

    if (wave == 0 && s + 2 < NSTAGES)    // refill freed buffer ASAP (overlaps WMMA)
      tdm_load_tile((unsigned)(size_t)(void*)&xbuf[s & 1][0], xrow + (s + 2) * KC);

    // ---- WMMA over the staged bf16 fragments ----
    #pragma unroll
    for (int kk = 0; kk < CPS; ++kk) {
      union { uint4 q[2]; v16bf v; } A;
      const uint4* af = (const uint4*)&abuf[(kk * 32 + lane) * 16];
      A.q[0] = af[0]; A.q[1] = af[1];

      const int c = s * CPS + kk;
      union { uint4 q[2]; v16bf v; } B0, B1;
      const uint4* b0p = (const uint4*)(wfrag + ((size_t)(t0 * KCHUNKS + c) * 32 + lane) * 16);
      const uint4* b1p = (const uint4*)(wfrag + ((size_t)(t1 * KCHUNKS + c) * 32 + lane) * 16);
      B0.q[0] = b0p[0]; B0.q[1] = b0p[1];
      B1.q[0] = b1p[0]; B1.q[1] = b1p[1];

      c0 = __builtin_amdgcn_wmma_f32_16x16x32_bf16(false, A.v, false, B0.v,
                                                   (short)0, c0, false, false);
      c1 = __builtin_amdgcn_wmma_f32_16x16x32_bf16(false, A.v, false, B1.v,
                                                   (short)0, c1, false, false);
    }
    // next iteration's first __syncthreads separates consumption from re-conversion
  }

  // Spill C fragments (16x16 f32 layout: VGPR r -> M = r + 8*(lane>=16), N = lane&15).
  {
    const int nb0 = t0 * 16 + (lane & 15);
    const int nb1 = t1 * 16 + (lane & 15);
    const int mo  = (lane >> 4) * 8;
    #pragma unroll
    for (int r = 0; r < 8; ++r) {
      logits[(mo + r) * NEXP + nb0] = c0[r];
      logits[(mo + r) * NEXP + nb1] = c1[r];
    }
  }
  __syncthreads();

  // Softmax: 16 lanes per token, half-wave shfl_xor reductions.
  float* scores = &xbuf[0][0];           // reuse TDM buffers (16x256 f32)
  {
    const int m = tid >> 4, su = tid & 15;
    const float* lrow = &logits[m * NEXP];
    float mx = -3.4e38f;
    #pragma unroll
    for (int i = 0; i < 16; ++i) mx = fmaxf(mx, lrow[su + 16 * i]);
    for (int d = 8; d >= 1; d >>= 1) mx = fmaxf(mx, __shfl_xor(mx, d, 16));
    float ev[16]; float sum = 0.f;
    #pragma unroll
    for (int i = 0; i < 16; ++i) { ev[i] = __expf(lrow[su + 16 * i] - mx); sum += ev[i]; }
    for (int d = 8; d >= 1; d >>= 1) sum += __shfl_xor(sum, d, 16);
    const float inv = 1.f / sum;
    #pragma unroll
    for (int i = 0; i < 16; ++i) scores[m * NEXP + su + 16 * i] = ev[i] * inv;
  }
  __syncthreads();

  // Grouped routing: one lane per token (matches jax top_k tie-breaking: strict >,
  // ascending scan => lowest index wins ties; selection emitted in descending order).
  if (tid < MTILE) {
    const int m = tid;
    const float* sc = &scores[m * NEXP];

    float gs[NGROUPS];
    #pragma unroll
    for (int g = 0; g < NGROUPS; ++g) {          // group score = top-2 sum of s+bias
      float m1 = -3.4e38f, m2 = -3.4e38f;
      for (int j = 0; j < GSIZE; ++j) {
        float v = sc[g * GSIZE + j] + sbias[g * GSIZE + j];
        if (v > m1) { m2 = m1; m1 = v; } else if (v > m2) { m2 = v; }
      }
      gs[g] = m1 + m2;
    }
    unsigned keep = 0;                           // top-4 groups
    for (int t = 0; t < TOPKG; ++t) {
      int bg = 0; float bv = -3.4e38f;
      for (int g = 0; g < NGROUPS; ++g)
        if (!((keep >> g) & 1) && gs[g] > bv) { bv = gs[g]; bg = g; }
      keep |= 1u << bg;
    }
    int idx[TOPK]; float wt[TOPK];               // top-8 experts in kept groups
    for (int t = 0; t < TOPK; ++t) {
      int be = 0; float bv = -3.4e38f;
      for (int g = 0; g < NGROUPS; ++g) {
        if (!((keep >> g) & 1)) continue;
        for (int j = 0; j < GSIZE; ++j) {
          int e = g * GSIZE + j;
          bool used = false;
          for (int u = 0; u < t; ++u) used |= (idx[u] == e);
          if (used) continue;
          float v = sc[e] + sbias[e];
          if (v > bv) { bv = v; be = e; }
        }
      }
      idx[t] = be;
      wt[t]  = sc[be] * ROUTE_SCALE;             // weight from original scores
    }
    const int row = mbase + m;
    #pragma unroll
    for (int t = 0; t < TOPK; ++t) {
      outw[row * TOPK + t] = wt[t];
      outi[row * TOPK + t] = (float)idx[t];
    }
  }
}

extern "C" void kernel_launch(void* const* d_in, const int* in_sizes, int n_in,
                              void* d_out, int out_size, void* d_ws, size_t ws_size,
                              hipStream_t stream) {
  const float* x    = (const float*)d_in[0];
  const float* w    = (const float*)d_in[1];
  const float* bias = (const float*)d_in[2];
  const int ntok = in_sizes[0] / DIM;

  unsigned short* wfrag = (unsigned short*)d_ws;        // 256*7168*2 = 3.67 MB

  swizzle_weight<<<(NEXP * DIM) / 256, 256, 0, stream>>>(w, wfrag);

  float* outw = (float*)d_out;
  float* outi = outw + (size_t)ntok * TOPK;
  gate_fused<<<ntok / MTILE, 256, 0, stream>>>(x, bias, wfrag, outw, outi);
}
